// ReBertSelfAttention_19052474925389
// MI455X (gfx1250) — matrix operand order repaired and checked by reading, not compiled, not run on hardware
//
#include <hip/hip_runtime.h>

#define D_MODEL 1024
#define HEADS 16
#define KV_HEADS 4
#define HEAD_DIM 64
#define SEQ 2048
#define BATCH 2
#define ROWS (BATCH * SEQ)          // 4096
#define KV_W (KV_HEADS * HEAD_DIM)  // 256

typedef __attribute__((ext_vector_type(16))) __bf16 v16bf;
typedef __attribute__((ext_vector_type(8)))  __bf16 v8bf;
typedef __attribute__((ext_vector_type(8)))  float  v8f;

union U32x8 { unsigned u[8]; v16bf b; };
union BFBits { __bf16 h; unsigned short s; };

__device__ __forceinline__ unsigned short bf_bits(float x) {
  BFBits t; t.h = (__bf16)x; return t.s;
}

__device__ __forceinline__ v8f v8f_zero() {
  v8f z = {0.f,0.f,0.f,0.f,0.f,0.f,0.f,0.f};
  return z;
}

__device__ __forceinline__ v8f wmma_bf16(v16bf a, v16bf b, v8f c) {
  return __builtin_amdgcn_wmma_f32_16x16x32_bf16(false, a, false, b, (short)0, c,
                                                 false, false);
}

// ---- WMMA fragment loaders: uniform base + 32-bit offsets, constexpr strides ----

// A fragment: rows row0..row0+15, K-dims kb..kb+31 of row-major [*, LD] bf16.
// A layout: lane&15 = M; lanes 0-15 K {0..15}, lanes 16-31 K {16..31} (contiguous, matches B loaders).
template <int LD>
__device__ __forceinline__ v16bf load_A_t(const __bf16* __restrict__ base,
                                          unsigned row0, unsigned kb) {
  const unsigned lane = threadIdx.x & 31;
  unsigned off = (row0 + (lane & 15)) * (unsigned)LD + kb + ((lane & 16) ? 16u : 0u);
  v8bf lo = *(const v8bf*)(base + off);
  v8bf hi = *(const v8bf*)(base + off + 8);
  return __builtin_shufflevector(lo, hi, 0,1,2,3,4,5,6,7,8,9,10,11,12,13,14,15);
}

// B fragment from pair-packed dwords Wp[K/2][N] (dword = {K even lo16, K odd hi16}).
template <int N>
__device__ __forceinline__ v16bf load_Bp_t(const unsigned* __restrict__ Wp,
                                           unsigned kb, unsigned nb) {
  const unsigned lane = threadIdx.x & 31;
  unsigned off = ((kb >> 1) + ((lane & 16) ? 8u : 0u)) * (unsigned)N + nb + (lane & 15);
  U32x8 r;
#pragma unroll
  for (int j = 0; j < 8; ++j) r.u[j] = Wp[off + j * N];
  return r.b;
}

// B fragment for K^T: B(dd,key) = Kmat[key0+key][col0+dd]; pairs of dd contiguous.
template <int LD>
__device__ __forceinline__ v16bf load_Bkt_t(const __bf16* __restrict__ Kmat,
                                            unsigned key0, unsigned col0) {
  const unsigned lane = threadIdx.x & 31;
  unsigned off = (key0 + (lane & 15)) * (unsigned)LD + col0 + ((lane & 16) ? 16u : 0u);
  U32x8 r;
#pragma unroll
  for (int j = 0; j < 8; ++j) r.u[j] = *(const unsigned*)(Kmat + off + 2 * j);
  return r.b;
}

// ---------------- Stage 1: conversions / packing ----------------

__global__ void cvt_bf16_kernel(const float* __restrict__ x,
                                unsigned short* __restrict__ y, int n) {
  int i = blockIdx.x * blockDim.x + threadIdx.x;
  if (i < n) y[i] = bf_bits(x[i]);
}

__global__ void pack_w_kernel(const float* __restrict__ W, unsigned* __restrict__ Wp,
                              int N, int n) {
  int i = blockIdx.x * blockDim.x + threadIdx.x;
  if (i >= n) return;
  int nn = i % N, k2 = i / N;
  unsigned lo = bf_bits(W[(size_t)(2 * k2) * N + nn]);
  unsigned hi = bf_bits(W[(size_t)(2 * k2 + 1) * N + nn]);
  Wp[i] = lo | (hi << 16);
}

// ---------------- Stage 2: projection GEMM (bf16 WMMA, pipelined) ----------------
template <int N>
__global__ void __launch_bounds__(256) gemm_kernel(const unsigned short* __restrict__ Au,
                                                   const unsigned* __restrict__ Wp,
                                                   const float* __restrict__ bias,
                                                   unsigned short* __restrict__ Cu) {
  const __bf16* A = (const __bf16*)Au;
  const unsigned lane = threadIdx.x & 31;
  int wv = blockIdx.x * (blockDim.x >> 5) + (threadIdx.x >> 5);
  const int tilesN = N >> 4;
  int mt = wv / tilesN, nt = wv % tilesN;
  const unsigned arow = mt * 16, nb = nt * 16;

  v8f acc = v8f_zero();
  v16bf a0 = load_A_t<D_MODEL>(A, arow, 0);
  v16bf b0 = load_Bp_t<N>(Wp, 0, nb);
#pragma unroll 4
  for (int kb = 0; kb < D_MODEL - 32; kb += 32) {
    v16bf a1 = load_A_t<D_MODEL>(A, arow, kb + 32);   // prefetch next step
    v16bf b1 = load_Bp_t<N>(Wp, kb + 32, nb);
    acc = wmma_bf16(a0, b0, acc);
    a0 = a1; b0 = b1;
  }
  acc = wmma_bf16(a0, b0, acc);

  unsigned col = nb + (lane & 15);
  float bv = bias[col];
  unsigned r0 = arow + ((lane & 16) ? 8u : 0u);
#pragma unroll
  for (int r = 0; r < 8; ++r)
    Cu[(size_t)(r0 + r) * N + col] = bf_bits(acc[r] + bv);
}

// ---------------- Stage 3: RoPE (rotate-half) ----------------
__global__ void rope_kernel(const unsigned short* __restrict__ xinu,
                            unsigned short* __restrict__ xoutu,
                            const int* __restrict__ pos_ids, int nheads, int n) {
  const __bf16* xin = (const __bf16*)xinu;
  int idx = blockIdx.x * blockDim.x + threadIdx.x;
  if (idx >= n) return;
  int i = idx & 31;
  int t = idx >> 5;
  int h = t % nheads;
  int row = t / nheads;                    // b*SEQ + s
  float pos = (float)pos_ids[row];
  float ang = pos * __expf(-(float)i * 0.2878231366242557f); // ln(1e4)/32
  float c = __cosf(ang), s = __sinf(ang);
  size_t base = (size_t)row * (nheads * HEAD_DIM) + h * HEAD_DIM;
  float x0 = (float)xin[base + i];
  float x1 = (float)xin[base + i + 32];
  xoutu[base + i]      = bf_bits(x0 * c - x1 * s);
  xoutu[base + i + 32] = bf_bits(x1 * c + x0 * s);
}

// ---------------- Stage 4: pack V into key-pair dwords ----------------
__global__ void pack_v_kernel(const unsigned short* __restrict__ vraw,
                              unsigned* __restrict__ vp, int n) {
  int idx = blockIdx.x * blockDim.x + threadIdx.x;
  if (idx >= n) return;
  int d = idx & 63;
  int t = idx >> 6;
  int s2 = t & (SEQ / 2 - 1);
  t >>= 10;
  int kvh = t & (KV_HEADS - 1);
  int b = t >> 2;
  size_t r0 = (size_t)(b * SEQ + 2 * s2) * KV_W + kvh * HEAD_DIM + d;
  vp[idx] = (unsigned)vraw[r0] | ((unsigned)vraw[r0 + KV_W] << 16);
}

// ---------------- Stage 5: flash attention (async-LDS staged K/V) ----------------
// Block = 4 waves = 4 consecutive q-tiles of the SAME (b,h): K/V tiles shared in LDS.
#define NW 4
__global__ void __launch_bounds__(NW * 32) attn_kernel(
    const unsigned short* __restrict__ Qu, const unsigned short* __restrict__ Ku,
    const unsigned* __restrict__ Vp, const float* __restrict__ mask,
    float* __restrict__ out) {
  const __bf16* Q = (const __bf16*)Qu;
  const __bf16* Kr = (const __bf16*)Ku;

  __shared__ __attribute__((aligned(16))) __bf16   kt[2][32][64];   // 8 KiB
  __shared__ __attribute__((aligned(16))) unsigned vt[2][16][64];   // 8 KiB
  __shared__ __attribute__((aligned(16))) float    sc[NW][16][33];
  __shared__ __attribute__((aligned(16))) __bf16   pr[NW][16][32];
  __shared__ float rowmax[NW][16], rowsum[NW][16], rscale[NW][16];

  const int tid = threadIdx.x;
  const int lane = tid & 31, wid = tid >> 5;
  int qt4 = blockIdx.x & (SEQ / 64 - 1);   // 32 per (b,h)
  int bh  = blockIdx.x >> 5;
  int h = bh & (HEADS - 1), b = bh >> 4;
  int kvh = h >> 2;                        // GQA: repeat_interleave => h/GROUPS
  int qt = qt4 * 4 + wid;

  const char* kgbase = (const char*)Kr + (size_t)(b * SEQ) * (KV_W * 2)
                                       + kvh * (HEAD_DIM * 2);
  const char* vgbase = (const char*)Vp
      + (size_t)(b * KV_HEADS + kvh) * (SEQ / 2) * (HEAD_DIM * 4);

  // Stage one 32-key chunk (K tile 32x64 bf16, packed-V tile 16x64 dw) into LDS.
  // 128 threads x 2 iters x 16B covers each 4 KiB tile; 4 async ops per lane.
  auto stage = [&](int buf, int jb) {
#pragma unroll
    for (int i = 0; i < 2; ++i) {
      int li = i * 128 + tid;
      int krow = li >> 3, kch = li & 7;
      unsigned klds = (unsigned)(size_t)(const void*)&kt[buf][krow][kch * 8];
      unsigned long long kga =
          (unsigned long long)(size_t)(kgbase + (size_t)(jb + krow) * (KV_W * 2) + kch * 16);
      asm volatile("global_load_async_to_lds_b128 %0, %1, off"
                   :: "v"(klds), "v"(kga) : "memory");
      int vrow = li >> 4, vch = li & 15;
      unsigned vlds = (unsigned)(size_t)(const void*)&vt[buf][vrow][vch * 4];
      unsigned long long vga =
          (unsigned long long)(size_t)(vgbase + (size_t)((jb >> 1) + vrow) * (HEAD_DIM * 4) + vch * 16);
      asm volatile("global_load_async_to_lds_b128 %0, %1, off"
                   :: "v"(vlds), "v"(vga) : "memory");
    }
  };

  if (lane < 16) { rowmax[wid][lane] = -1e30f; rowsum[wid][lane] = 0.f; }

  v8f acc[4];
#pragma unroll
  for (int t = 0; t < 4; ++t) acc[t] = v8f_zero();

  // Q fragments (head-dim 0..31 and 32..63) stay resident in VGPRs.
  v16bf qa0 = load_A_t<D_MODEL>(Q, b * SEQ + qt * 16, h * HEAD_DIM);
  v16bf qa1 = load_A_t<D_MODEL>(Q, b * SEQ + qt * 16, h * HEAD_DIM + 32);

  const int n15 = lane & 15;
  const int rb = (lane & 16) ? 8 : 0;
  const size_t mbase = ((size_t)b * SEQ + qt * 16) * SEQ;

  // One 32-key chunk of online-softmax attention from LDS tiles kt/vt[buf].
  auto compute = [&](int buf, int jb) {
    // --- scores: two 16x16 tiles over keys jb..jb+31, from LDS K tile ---
    const __bf16* kl = &kt[buf][0][0];
    v8f s0 = v8f_zero(), s1 = v8f_zero();
    s0 = wmma_bf16(qa0, load_Bkt_t<64>(kl, 0, 0), s0);
    s0 = wmma_bf16(qa1, load_Bkt_t<64>(kl, 0, 32), s0);
    s1 = wmma_bf16(qa0, load_Bkt_t<64>(kl, 16, 0), s1);
    s1 = wmma_bf16(qa1, load_Bkt_t<64>(kl, 16, 32), s1);

#pragma unroll
    for (int r = 0; r < 8; ++r) {
      size_t mrow = mbase + (size_t)(rb + r) * SEQ;
      sc[wid][rb + r][n15]      = s0[r] * 0.125f + mask[mrow + jb + n15];
      sc[wid][rb + r][16 + n15] = s1[r] * 0.125f + mask[mrow + jb + 16 + n15];
      __builtin_prefetch(&mask[mrow + jb + 32 + n15], 0, 1); // speculative, safe
    }
    __syncthreads();

    // --- online softmax, one row per lane (lanes 0-15) ---
    if (lane < 16) {
      float mprev = rowmax[wid][lane];
      float mx = mprev;
#pragma unroll
      for (int j = 0; j < 32; ++j) mx = fmaxf(mx, sc[wid][lane][j]);
      float scale = __expf(mprev - mx);
      float sum = rowsum[wid][lane] * scale;
#pragma unroll
      for (int j = 0; j < 32; ++j) {
        float e = __expf(sc[wid][lane][j] - mx);
        sum += e;
        pr[wid][lane][j] = (__bf16)e;
      }
      rowmax[wid][lane] = mx;
      rowsum[wid][lane] = sum;
      rscale[wid][lane] = scale;
    }
    __syncthreads();

    // --- rescale accumulators (per-row factor) ---
    float f[8];
#pragma unroll
    for (int r = 0; r < 8; ++r) f[r] = rscale[wid][rb + r];
#pragma unroll
    for (int t = 0; t < 4; ++t)
#pragma unroll
      for (int r = 0; r < 8; ++r) acc[t][r] *= f[r];

    // --- PV: probs (LDS, A-layout) x V (LDS, packed B-layout) ---
    v16bf pa = load_A_t<32>((const __bf16*)&pr[wid][0][0], 0, 0);
    const unsigned* vl = &vt[buf][0][0];
#pragma unroll
    for (int t = 0; t < 4; ++t)
      acc[t] = wmma_bf16(pa, load_Bp_t<64>(vl, 0, t * 16), acc[t]);
    __syncthreads();
  };

  // Software-pipelined main loop (branch-free steady state) + peeled tail.
  stage(0, 0);
  for (int jb = 0; jb < SEQ - 32; jb += 32) {
    int buf = (jb >> 5) & 1;
    stage(buf ^ 1, jb + 32);                          // overlap next chunk
    asm volatile("s_wait_asynccnt 0x4" ::: "memory"); // current 4 done, next 4 fly
    __syncthreads();  // all waves' portions of kt/vt[buf] are in LDS
    compute(buf, jb);
  }
  asm volatile("s_wait_asynccnt 0x0" ::: "memory");
  __syncthreads();
  compute(((SEQ - 32) >> 5) & 1, SEQ - 32);

  // --- epilogue: divide by running sum, write [B,S,H*hd] ---
#pragma unroll
  for (int r = 0; r < 8; ++r) {
    float inv = 1.0f / rowsum[wid][rb + r];
    size_t orow = (size_t)(b * SEQ + qt * 16 + rb + r) * D_MODEL + h * HEAD_DIM;
#pragma unroll
    for (int t = 0; t < 4; ++t)
      out[orow + t * 16 + n15] = acc[t][r] * inv;
  }
}

// ---------------- host ----------------
extern "C" void kernel_launch(void* const* d_in, const int* in_sizes, int n_in,
                              void* d_out, int out_size, void* d_ws, size_t ws_size,
                              hipStream_t stream) {
  const float* seq  = (const float*)d_in[0];
  const float* mask = (const float*)d_in[1];
  const int*   pos  = (const int*)d_in[2];
  const float* Wq   = (const float*)d_in[3];
  const float* bq   = (const float*)d_in[4];
  const float* Wk   = (const float*)d_in[5];
  const float* bk   = (const float*)d_in[6];
  const float* Wv   = (const float*)d_in[7];
  const float* bv   = (const float*)d_in[8];
  float* out = (float*)d_out;

  char* ws = (char*)d_ws;
  // 256-aligned workspace layout (bytes)
  unsigned short* seqb  = (unsigned short*)(ws + 0);          // 4096*1024 bf16 = 8 MiB
  unsigned*       wqp   = (unsigned*)(ws + 8388608);          // 512*1024 dw   = 2 MiB
  unsigned*       wkp   = (unsigned*)(ws + 10485760);         // 512*256 dw
  unsigned*       wvp   = (unsigned*)(ws + 11010048);         // 512*256 dw
  unsigned short* qraw  = (unsigned short*)(ws + 11534336);   // 4096*1024 bf16
  unsigned short* kraw  = (unsigned short*)(ws + 19922944);   // 4096*256 bf16
  unsigned short* vraw  = (unsigned short*)(ws + 22020096);   // 4096*256 bf16
  unsigned short* qrope = (unsigned short*)(ws + 24117248);   // 4096*1024 bf16
  unsigned short* krope = (unsigned short*)(ws + 32505856);   // 4096*256 bf16
  unsigned*       vpk   = (unsigned*)(ws + 34603008);         // 2*4*1024*64 dw

  // Stage 1: convert + pack
  cvt_bf16_kernel<<<(ROWS * D_MODEL) / 256, 256, 0, stream>>>(seq, seqb, ROWS * D_MODEL);
  pack_w_kernel<<<(512 * 1024) / 256, 256, 0, stream>>>(Wq, wqp, 1024, 512 * 1024);
  pack_w_kernel<<<(512 * 256) / 256, 256, 0, stream>>>(Wk, wkp, 256, 512 * 256);
  pack_w_kernel<<<(512 * 256) / 256, 256, 0, stream>>>(Wv, wvp, 256, 512 * 256);

  // Stage 2: projections (one wave per 16x16 tile, 8 waves per block)
  gemm_kernel<1024><<<(256 * 64) / 8, 256, 0, stream>>>(seqb, wqp, bq, qraw);
  gemm_kernel<256><<<(256 * 16) / 8, 256, 0, stream>>>(seqb, wkp, bk, kraw);
  gemm_kernel<256><<<(256 * 16) / 8, 256, 0, stream>>>(seqb, wvp, bv, vraw);

  // Stage 3: RoPE on Q and K
  rope_kernel<<<(ROWS * HEADS * 32) / 256, 256, 0, stream>>>(qraw, qrope, pos, HEADS,
                                                             ROWS * HEADS * 32);
  rope_kernel<<<(ROWS * KV_HEADS * 32) / 256, 256, 0, stream>>>(kraw, krope, pos,
                                                                KV_HEADS,
                                                                ROWS * KV_HEADS * 32);

  // Stage 4: pack V
  pack_v_kernel<<<(BATCH * KV_HEADS * (SEQ / 2) * HEAD_DIM) / 256, 256, 0, stream>>>(
      vraw, vpk, BATCH * KV_HEADS * (SEQ / 2) * HEAD_DIM);

  // Stage 5: attention — one block per (b,h,64 queries), 4 waves
  attn_kernel<<<BATCH * HEADS * (SEQ / 64), NW * 32, 0, stream>>>(
      qrope, krope, vpk, mask, out);
}
